// ProgSGMultimodalModel_8821862826774
// MI455X (gfx1250) — compile-verified, matrix operand order checked by reading, not run on hardware
//
#include <hip/hip_runtime.h>
#include <hip/hip_bf16.h>
#include <math.h>

#define NNODE 30000
#define NEDGE 100000
#define NB    16
#define HD    512
#define NHEAD 8
#define CHD   64
#define LCONV 8
#define LTOK  512
#define CODED 768
#define GATEH 256
#define NT    4
#define BTOK  (NB*LTOK) /* 8192 */

typedef _Float16 f16;
typedef f16  v16h __attribute__((ext_vector_type(16)));
typedef f16  v8h  __attribute__((ext_vector_type(8)));
typedef float v8f __attribute__((ext_vector_type(8)));

// ---------------------------------------------------------------------------
// WMMA fragment loader: 16-bit A/B fragment for v_wmma_f32_16x16x32_f16.
// Lane l (group g=l>>4) holds K = {kk+8g .. kk+8g+7} and {kk+16+8g .. +7},
// i.e. two contiguous 16-byte chunks of a [row, K] matrix (compiler merges
// them into global_load_b128 pairs).
// ---------------------------------------------------------------------------
__device__ __forceinline__ v16h load_frag(const f16* __restrict__ p) {
  v8h lo = *(const v8h*)(p);
  v8h hi = *(const v8h*)(p + 16);
  v16h r;
#pragma unroll
  for (int i = 0; i < 8; ++i) { r[i] = lo[i]; r[i + 8] = hi[i]; }
  return r;
}

__device__ __forceinline__ float actf(float v, int act) {
  if (act == 1) return v > 0.f ? v : (expf(v) - 1.f);   // ELU
  if (act == 2) return v > 0.f ? v : 0.f;               // ReLU
  return v;
}

// ---------------------------------------------------------------------------
// Generic batched GEMM: C[z] = act(scale * A[z] @ W[z]^T + bias[z])
// A: [M,K] f16 (row stride == K), W: [Nn,K] f16, bias f32 (or null).
// C offset per z: (z/zdiv)*sC1 + (z%zdiv)*sC2, row stride ldc.
// Block = 256 threads = 8 wave32; block tile 16x512, per-wave 16x64 via
// 4 WMMAs sharing one A fragment (A streamed from HBM once when Nn<=512).
// Requires M%16==0, K%32==0, Nn%64==0.
// ---------------------------------------------------------------------------
__global__ __launch_bounds__(256)
void k_gemm(const f16* __restrict__ A, const f16* __restrict__ W,
            const float* __restrict__ bias, void* __restrict__ Cout,
            int M, int Nn, int K, int ldc,
            long long sA, long long sW, long long sC1, long long sC2,
            int zdiv, float scale, int act, int biasStride, int outHalf) {
  int z = blockIdx.z;
  const f16* Ab = A + (size_t)z * (size_t)sA;
  const f16* Wb = W + (size_t)z * (size_t)sW;
  size_t offC = (size_t)(z / zdiv) * (size_t)sC1 + (size_t)(z % zdiv) * (size_t)sC2;
  int lane = threadIdx.x & 31, wave = threadIdx.x >> 5;
  int tileM = blockIdx.x << 4;
  int tileN = (blockIdx.y << 9) + (wave << 6);
  if (tileN >= Nn) return;                 // wave-uniform: EXEC stays all-ones
  int g = lane >> 4, r0 = lane & 15;
  const f16* pa = Ab + (size_t)(tileM + r0) * K + g * 8;
  const f16* pw = Wb + (size_t)(tileN + r0) * K + g * 8;
  size_t wrow16 = (size_t)16 * K;          // 16 W rows
  v8f acc[4];
#pragma unroll
  for (int j = 0; j < 4; ++j)
#pragma unroll
    for (int r = 0; r < 8; ++r) acc[j][r] = 0.f;
  for (int kk = 0; kk < K; kk += 32) {
    v16h a = load_frag(pa + kk);
#pragma unroll
    for (int j = 0; j < 4; ++j) {
      v16h b = load_frag(pw + (size_t)j * wrow16 + kk);
      acc[j] = __builtin_amdgcn_wmma_f32_16x16x32_f16(false, a, false, b, (short)0,
                                                      acc[j], false, false);
    }
  }
#pragma unroll
  for (int j = 0; j < 4; ++j) {
    int n = tileN + j * 16 + r0;
    float bv = bias ? bias[(size_t)z * biasStride + n] : 0.f;
#pragma unroll
    for (int r = 0; r < 8; ++r) {
      int m = tileM + r + (g << 3);
      float v = actf(acc[j][r] * scale + bv, act);
      size_t i0 = offC + (size_t)m * ldc + n;
      if (outHalf) ((f16*)Cout)[i0] = (f16)v;
      else         ((float*)Cout)[i0] = v;
    }
  }
}

// ------------------------------ helpers ------------------------------------
__device__ __forceinline__ void atomicMaxF(float* addr, float val) {
  unsigned int old = *(volatile unsigned int*)addr;
  while (__uint_as_float(old) < val) {
    unsigned int assumed = old;
    old = atomicCAS((unsigned int*)addr, assumed, __float_as_uint(val));
    if (old == assumed) break;
  }
}

__global__ void k_cvt(const float* __restrict__ s, f16* __restrict__ d, long long n) {
  long long i = (long long)blockIdx.x * 256 + threadIdx.x;
  if (i < n) d[i] = (f16)s[i];
}

__global__ void k_zero_f(float* p, long long n) {
  long long i = (long long)blockIdx.x * 256 + threadIdx.x;
  if (i < n) p[i] = 0.f;
}
__global__ void k_zero_d(double* p, int n) {
  int i = blockIdx.x * 64 + threadIdx.x;
  if (i < n) p[i] = 0.0;
}
__global__ void k_init_ms(float* m, float* s, int n) {
  int i = blockIdx.x * 256 + threadIdx.x;
  if (i < n) { m[i] = -__builtin_inff(); s[i] = 0.f; }
}
__global__ void k_fixm(float* m, int n) {
  int i = blockIdx.x * 256 + threadIdx.x;
  if (i < n) { float v = m[i]; if (!isfinite(v)) m[i] = 0.f; }
}

// ------------------------------ embeddings ---------------------------------
__global__ __launch_bounds__(512)
void k_embed_nodes(const int* __restrict__ x, const float* __restrict__ emb,
                   float* __restrict__ h, f16* __restrict__ h16) {
  int n = blockIdx.x, d = threadIdx.x;
  float s = 0.f;
#pragma unroll
  for (int f = 0; f < 4; ++f) {
    int idx = x[n * 4 + f];
    s += emb[((size_t)f * 64 + idx) * HD + d];
  }
  size_t i = (size_t)n * HD + d;
  h[i] = s; h16[i] = (f16)s;
}

__global__ __launch_bounds__(512)
void k_embed_edges(const int* __restrict__ ea, const float* __restrict__ emb,
                   f16* __restrict__ ef16) {
  int e = blockIdx.x, d = threadIdx.x;
  float s = 0.f;
#pragma unroll
  for (int f = 0; f < 2; ++f) {
    int idx = ea[e * 2 + f];
    s += emb[((size_t)f * 16 + idx) * HD + d];
  }
  ef16[(size_t)e * HD + d] = (f16)s;
}

// ------------------------------ graph attention ----------------------------
__global__ void k_logits(const f16* __restrict__ q, const f16* __restrict__ k,
                         const f16* __restrict__ ep, const int* __restrict__ src,
                         const int* __restrict__ dst, float* __restrict__ logits) {
  int i = blockIdx.x * 256 + threadIdx.x;
  if (i >= NEDGE * NHEAD) return;
  int e = i >> 3, hh = i & 7;
  const f16* qp = q + (size_t)dst[e] * HD + hh * CHD;
  const f16* kp = k + (size_t)src[e] * HD + hh * CHD;
  const f16* pp = ep + (size_t)e * HD + hh * CHD;
  float acc = 0.f;
  for (int c = 0; c < CHD; ++c)
    acc += (float)qp[c] * ((float)kp[c] + (float)pp[c]);
  logits[i] = acc * 0.125f;   // 1/sqrt(64)
}

__global__ void k_segmax_e(const float* __restrict__ lg, const int* __restrict__ dst,
                           float* __restrict__ m) {
  int i = blockIdx.x * 256 + threadIdx.x;
  if (i >= NEDGE * NHEAD) return;
  atomicMaxF(&m[dst[i >> 3] * NHEAD + (i & 7)], lg[i]);
}

__global__ void k_expsum_e(float* __restrict__ lg, const int* __restrict__ dst,
                           const float* __restrict__ m, float* __restrict__ s) {
  int i = blockIdx.x * 256 + threadIdx.x;
  if (i >= NEDGE * NHEAD) return;
  int sg = dst[i >> 3] * NHEAD + (i & 7);
  float ev = expf(lg[i] - m[sg]);
  lg[i] = ev;
  atomicAdd(&s[sg], ev);
}

__global__ void k_scatter_agg(const float* __restrict__ ee, const float* __restrict__ s,
                              const f16* __restrict__ v, const f16* __restrict__ ep,
                              const int* __restrict__ src, const int* __restrict__ dst,
                              float* __restrict__ agg) {
  long long i = (long long)blockIdx.x * 256 + threadIdx.x;
  if (i >= (long long)NEDGE * HD) return;
  int e = (int)(i >> 9), d = (int)(i & 511), hh = d >> 6;
  int dN = dst[e];
  float alpha = ee[e * NHEAD + hh] / (s[dN * NHEAD + hh] + 1e-16f);
  float val = ((float)v[(size_t)src[e] * HD + d] + (float)ep[(size_t)e * HD + d]) * alpha;
  atomicAdd(&agg[(size_t)dN * HD + d], val);
}

__global__ __launch_bounds__(256)
void k_beta(const float* __restrict__ agg, const float* __restrict__ xr,
            const float* __restrict__ wb, float* __restrict__ beta) {
  __shared__ float red[256];
  int n = blockIdx.x, t = threadIdx.x;
  float p = 0.f;
  for (int d = t; d < HD; d += 256) {
    float a = agg[(size_t)n * HD + d], x = xr[(size_t)n * HD + d];
    p += a * wb[d] + x * wb[HD + d] + (a - x) * wb[2 * HD + d];
  }
  red[t] = p; __syncthreads();
  for (int st = 128; st > 0; st >>= 1) { if (t < st) red[t] += red[t + st]; __syncthreads(); }
  if (t == 0) beta[n] = 1.f / (1.f + expf(-red[0]));
}

__global__ __launch_bounds__(256)
void k_hn(const float* __restrict__ agg, const float* __restrict__ xr,
          const float* __restrict__ beta, float* __restrict__ hn, double* red) {
  __shared__ float r1[256], r2[256];
  long long i = (long long)blockIdx.x * 256 + threadIdx.x;
  float val = 0.f;
  if (i < (long long)NNODE * HD) {
    int n = (int)(i >> 9);
    float b = beta[n];
    float v = b * xr[i] + (1.f - b) * agg[i];
    v = v > 0.f ? v : (expf(v) - 1.f);   // ELU
    hn[i] = v;
    val = v;
  }
  int t = threadIdx.x;
  r1[t] = val; r2[t] = val * val; __syncthreads();
  for (int st = 128; st > 0; st >>= 1) {
    if (t < st) { r1[t] += r1[t + st]; r2[t] += r2[t + st]; } __syncthreads();
  }
  if (t == 0) { atomicAdd(&red[0], (double)r1[0]); atomicAdd(&red[1], (double)r2[0]); }
}

__global__ void k_update(const float* __restrict__ hn, float* __restrict__ h,
                         f16* __restrict__ h16, float* __restrict__ finalb,
                         const float* __restrict__ lnw, const float* __restrict__ lnb,
                         const double* __restrict__ red, int first) {
  long long i = (long long)blockIdx.x * 256 + threadIdx.x;
  if (i >= (long long)NNODE * HD) return;
  double cnt = (double)NNODE * HD;
  double m = red[0] / cnt;
  double var = red[1] / cnt - m * m;
  float stdv = sqrtf(fmaxf((float)var, 0.f));
  int d = (int)(i & 511);
  float val = ((hn[i] - (float)m) / (stdv + 1e-5f)) * lnw[d] + lnb[d];
  float hnew = h[i] + val;
  h[i] = hnew; h16[i] = (f16)hnew;
  finalb[i] = first ? hnew : fmaxf(finalb[i], hnew);
}

// ------------------------------ pooling ------------------------------------
__global__ __launch_bounds__(256)
void k_gate2(const f16* __restrict__ g1, const float* __restrict__ w2,
             const float* __restrict__ b2, float* __restrict__ gate) {
  __shared__ float red[256];
  int n = blockIdx.x, t = threadIdx.x;
  red[t] = (float)g1[(size_t)n * GATEH + t] * w2[t];
  __syncthreads();
  for (int st = 128; st > 0; st >>= 1) { if (t < st) red[t] += red[t + st]; __syncthreads(); }
  if (t == 0) gate[n] = red[0] + b2[0];
}

__global__ void k_segmax1(const float* __restrict__ v, const int* __restrict__ seg,
                          float* __restrict__ m, int n) {
  int i = blockIdx.x * 256 + threadIdx.x;
  if (i < n) atomicMaxF(&m[seg[i]], v[i]);
}
__global__ void k_expsum1(float* __restrict__ v, const int* __restrict__ seg,
                          const float* __restrict__ m, float* __restrict__ s, int n) {
  int i = blockIdx.x * 256 + threadIdx.x;
  if (i < n) { float e = expf(v[i] - m[seg[i]]); v[i] = e; atomicAdd(&s[seg[i]], e); }
}
__global__ void k_pool_scatter(const float* __restrict__ fin, const float* __restrict__ a,
                               const float* __restrict__ s, const int* __restrict__ seg,
                               float* __restrict__ pooled) {
  long long i = (long long)blockIdx.x * 256 + threadIdx.x;
  if (i >= (long long)NNODE * HD) return;
  int n = (int)(i >> 9), d = (int)(i & 511);
  int b = seg[n];
  atomicAdd(&pooled[b * HD + d], fin[i] * a[n] / (s[b] + 1e-16f));
}

__global__ void k_sctx1(const float* __restrict__ sc, const float* __restrict__ w1,
                        const float* __restrict__ b1, float* __restrict__ hid) {
  int i = blockIdx.x * 256 + threadIdx.x;
  if (i >= NB * HD) return;
  int b = i >> 9, j = i & 511;
  float acc = b1[j];
#pragma unroll
  for (int f = 0; f < 5; ++f) acc += sc[b * 5 + f] * w1[j * 5 + f];
  hid[i] = acc > 0.f ? acc : (expf(acc) - 1.f);
}
__global__ void k_sctx2(const float* __restrict__ hid, const float* __restrict__ w2,
                        const float* __restrict__ b2, const float* __restrict__ pooled,
                        float* __restrict__ gemb) {
  int i = blockIdx.x * 256 + threadIdx.x;
  if (i >= NB * HD) return;
  int b = i >> 9, d = i & 511;
  float acc = b2[d];
  const float* hp = hid + b * HD;
  const float* wp = w2 + (size_t)d * HD;
  for (int j = 0; j < HD; ++j) acc += hp[j] * wp[j];
  gemb[i] = pooled[i] + acc;
}

// ------------------------------ code branch --------------------------------
__global__ void k_qkv_split(const f16* __restrict__ qkv, f16* __restrict__ q16,
                            f16* __restrict__ k16, f16* __restrict__ vT16) {
  int i = blockIdx.x * 256 + threadIdx.x;
  if (i >= BTOK * HD) return;                     // 4,194,304
  int c  = i & 63;
  int hh = (i >> 6) & 7;
  int tk = (i >> 9) & 511;
  int b  = i >> 18;
  size_t src = ((size_t)(b * LTOK + tk)) * (3 * HD) + hh * CHD + c;
  int z = b * NHEAD + hh;
  size_t qd = ((size_t)z * LTOK + tk) * CHD + c;
  q16[qd] = qkv[src];
  k16[qd] = qkv[src + HD];
  vT16[((size_t)z * CHD + c) * LTOK + tk] = qkv[src + 2 * HD];
}

__global__ __launch_bounds__(256)
void k_softmax(const float* __restrict__ sc, f16* __restrict__ pr) {
  __shared__ float red[256];
  __shared__ float bc;
  int row = blockIdx.x, t = threadIdx.x;
  const float* p = sc + (size_t)row * 512;
  float a = p[t], b = p[t + 256];
  red[t] = fmaxf(a, b); __syncthreads();
  for (int st = 128; st > 0; st >>= 1) { if (t < st) red[t] = fmaxf(red[t], red[t + st]); __syncthreads(); }
  if (t == 0) bc = red[0];
  __syncthreads();
  float mx = bc;
  float e0 = expf(a - mx), e1 = expf(b - mx);
  __syncthreads();
  red[t] = e0 + e1; __syncthreads();
  for (int st = 128; st > 0; st >>= 1) { if (t < st) red[t] += red[t + st]; __syncthreads(); }
  if (t == 0) bc = red[0];
  __syncthreads();
  float inv = 1.f / bc;
  f16* o = pr + (size_t)row * 512;
  o[t] = (f16)(e0 * inv); o[t + 256] = (f16)(e1 * inv);
}

__global__ __launch_bounds__(256)
void k_lnres(float* __restrict__ t, const float* __restrict__ add,
             const float* __restrict__ w, const float* __restrict__ b,
             f16* __restrict__ t16) {
  __shared__ float red[256];
  __shared__ float bc;
  int row = blockIdx.x, tid = threadIdx.x;
  size_t base = (size_t)row * HD;
  float x0 = t[base + tid] + add[base + tid];
  float x1 = t[base + tid + 256] + add[base + tid + 256];
  red[tid] = x0 + x1; __syncthreads();
  for (int st = 128; st > 0; st >>= 1) { if (tid < st) red[tid] += red[tid + st]; __syncthreads(); }
  if (tid == 0) bc = red[0] * (1.f / HD);
  __syncthreads();
  float m = bc;
  float d0 = x0 - m, d1 = x1 - m;
  __syncthreads();
  red[tid] = d0 * d0 + d1 * d1; __syncthreads();
  for (int st = 128; st > 0; st >>= 1) { if (tid < st) red[tid] += red[tid + st]; __syncthreads(); }
  if (tid == 0) bc = red[0] * (1.f / HD);
  __syncthreads();
  float inv = rsqrtf(bc + 1e-5f);
  float o0 = d0 * inv * w[tid] + b[tid];
  float o1 = d1 * inv * w[tid + 256] + b[tid + 256];
  t[base + tid] = o0;         t[base + tid + 256] = o1;
  t16[base + tid] = (f16)o0;  t16[base + tid + 256] = (f16)o1;
}

__global__ void k_mean_tokens(const float* __restrict__ t, float* __restrict__ cemb) {
  int i = blockIdx.x * 256 + threadIdx.x;
  if (i >= NB * HD) return;
  int b = i >> 9, d = i & 511;
  float s = 0.f;
  const float* p = t + (size_t)b * LTOK * HD + d;
  for (int tk = 0; tk < LTOK; ++tk) s += p[(size_t)tk * HD];
  cemb[i] = s * (1.f / LTOK);
}

__global__ void k_fuse(const float* __restrict__ g, const float* __restrict__ c,
                       f16* __restrict__ fused) {
  int i = blockIdx.x * 256 + threadIdx.x;
  if (i >= NB * 2 * HD) return;
  int b = i >> 10, j = i & 1023;
  float v = (j < HD) ? g[b * HD + j] : c[b * HD + (j - HD)];
  fused[i] = (f16)v;
}

__global__ void k_head4(const f16* __restrict__ z3, const float* __restrict__ w4,
                        const float* __restrict__ b4, float* __restrict__ out) {
  int i = threadIdx.x;
  if (i >= NB * NT) return;
  int b = i >> 2, tt = i & 3;
  const f16* zp = z3 + ((size_t)tt * NB + b) * 128;
  const float* wp = w4 + tt * 128;
  float acc = b4[tt];
  for (int j = 0; j < 128; ++j) acc += (float)zp[j] * wp[j];
  out[b * NT + tt] = acc;     // zz[:, :, 0].T -> [B, NTARG]
}

// ------------------------------ host side ----------------------------------
static void gemm(hipStream_t st, const f16* A, const f16* W, const float* bias,
                 void* C, int M, int Nn, int K, int ldc,
                 long long sA, long long sW, long long sC1, long long sC2,
                 int zdiv, float scale, int act, int biasStride, int outHalf, int batch) {
  dim3 grid((unsigned)(M / 16), (unsigned)((Nn + 511) / 512), (unsigned)batch);
  k_gemm<<<grid, 256, 0, st>>>(A, W, bias, C, M, Nn, K, ldc, sA, sW, sC1, sC2,
                               zdiv, scale, act, biasStride, outHalf);
}

extern "C" void kernel_launch(void* const* d_in, const int* in_sizes, int n_in,
                              void* d_out, int out_size, void* d_ws, size_t ws_size,
                              hipStream_t stream) {
  (void)in_sizes; (void)n_in; (void)out_size; (void)ws_size;
  // ---- inputs ----
  const int*   x         = (const int*)d_in[0];
  const int*   esrc      = (const int*)d_in[1];
  const int*   edst      = esrc + NEDGE;
  const int*   edge_attr = (const int*)d_in[2];
  const int*   batch_vec = (const int*)d_in[3];
  const float* scalars   = (const float*)d_in[4];
  const float* code_tok  = (const float*)d_in[5];
  const float* node_emb  = (const float*)d_in[6];
  const float* edge_emb  = (const float*)d_in[7];
  const float* Wq = (const float*)d_in[8];   const float* bq = (const float*)d_in[9];
  const float* Wk = (const float*)d_in[10];  const float* bk = (const float*)d_in[11];
  const float* Wv = (const float*)d_in[12];  const float* bv = (const float*)d_in[13];
  const float* We = (const float*)d_in[14];
  const float* Wsk = (const float*)d_in[15]; const float* bsk = (const float*)d_in[16];
  const float* Wbeta = (const float*)d_in[17];
  const float* lnw = (const float*)d_in[18]; const float* lnb = (const float*)d_in[19];
  const float* gW1 = (const float*)d_in[20]; const float* gb1 = (const float*)d_in[21];
  const float* gW2 = (const float*)d_in[22]; const float* gb2 = (const float*)d_in[23];
  const float* sW1 = (const float*)d_in[24]; const float* sb1 = (const float*)d_in[25];
  const float* sW2 = (const float*)d_in[26]; const float* sb2 = (const float*)d_in[27];
  const float* cpW = (const float*)d_in[28]; const float* cpb = (const float*)d_in[29];
  const float* aWin = (const float*)d_in[30]; const float* abin = (const float*)d_in[31];
  const float* aWout = (const float*)d_in[32]; const float* about = (const float*)d_in[33];
  const float* fW1 = (const float*)d_in[34]; const float* fb1 = (const float*)d_in[35];
  const float* fW2 = (const float*)d_in[36]; const float* fb2 = (const float*)d_in[37];
  const float* t1w = (const float*)d_in[38]; const float* t1b = (const float*)d_in[39];
  const float* t2w = (const float*)d_in[40]; const float* t2b = (const float*)d_in[41];
  const float* hW1 = (const float*)d_in[42]; const float* hb1 = (const float*)d_in[43];
  const float* hW2 = (const float*)d_in[44]; const float* hb2 = (const float*)d_in[45];
  const float* hW3 = (const float*)d_in[46]; const float* hb3 = (const float*)d_in[47];
  const float* hW4 = (const float*)d_in[48]; const float* hb4 = (const float*)d_in[49];

  // ---- workspace bump allocator ----
  size_t off = 0;
  auto alloc = [&](size_t bytes) -> void* {
    void* p = (char*)d_ws + off;
    off = (off + bytes + 255) & ~((size_t)255);
    return p;
  };
  // persistent
  float* h      = (float*)alloc((size_t)NNODE * HD * 4);
  f16*   h16    = (f16*)  alloc((size_t)NNODE * HD * 2);
  f16*   ef16   = (f16*)  alloc((size_t)NEDGE * HD * 2);
  float* finalb = (float*)alloc((size_t)NNODE * HD * 4);
  float* logits = (float*)alloc((size_t)NEDGE * NHEAD * 4);
  float* segm   = (float*)alloc((size_t)NNODE * NHEAD * 4);
  float* segs   = (float*)alloc((size_t)NNODE * NHEAD * 4);
  float* beta   = (float*)alloc((size_t)NNODE * 4);
  double* red   = (double*)alloc(64);
  float* pm     = (float*)alloc(NB * 4);
  float* ps     = (float*)alloc(NB * 4);
  float* gateb  = (float*)alloc((size_t)NNODE * 4);
  float* pooled = (float*)alloc(NB * HD * 4);
  float* shid   = (float*)alloc(NB * HD * 4);
  float* gemb   = (float*)alloc(NB * HD * 4);
  float* cemb   = (float*)alloc(NB * HD * 4);
  f16* fused16  = (f16*)alloc(NB * 2 * HD * 2);
  f16* z1_16    = (f16*)alloc(NT * NB * HD * 2);
  f16* z2_16    = (f16*)alloc(NT * NB * 256 * 2);
  f16* z3_16    = (f16*)alloc(NT * NB * 128 * 2);
  // f16 weight copies
  f16* wq16  = (f16*)alloc((size_t)LCONV * HD * HD * 2);
  f16* wk16  = (f16*)alloc((size_t)LCONV * HD * HD * 2);
  f16* wv16  = (f16*)alloc((size_t)LCONV * HD * HD * 2);
  f16* we16  = (f16*)alloc((size_t)LCONV * HD * HD * 2);
  f16* wsk16 = (f16*)alloc((size_t)LCONV * HD * HD * 2);
  f16* gw116 = (f16*)alloc((size_t)GATEH * HD * 2);
  f16* cpw16 = (f16*)alloc((size_t)HD * CODED * 2);
  f16* win16 = (f16*)alloc((size_t)2 * 3 * HD * HD * 2);
  f16* wout16= (f16*)alloc((size_t)2 * HD * HD * 2);
  f16* fw116 = (f16*)alloc((size_t)2 * 4 * HD * HD * 2);
  f16* fw216 = (f16*)alloc((size_t)2 * HD * 4 * HD * 2);
  f16* hw116 = (f16*)alloc((size_t)NT * HD * 2 * HD * 2);
  f16* hw216 = (f16*)alloc((size_t)NT * 256 * HD * 2);
  f16* hw316 = (f16*)alloc((size_t)NT * 128 * 256 * 2);
  // shared scratch pool (phase-overlapped)
  size_t poolBase = off;
  // graph phase
  f16*   eproj16 = (f16*)  alloc((size_t)NEDGE * HD * 2);
  f16*   qb      = (f16*)  alloc((size_t)NNODE * HD * 2);
  f16*   kb      = (f16*)  alloc((size_t)NNODE * HD * 2);
  f16*   vb      = (f16*)  alloc((size_t)NNODE * HD * 2);
  float* xr      = (float*)alloc((size_t)NNODE * HD * 4);
  float* agg     = (float*)alloc((size_t)NNODE * HD * 4);
  float* hnbuf   = (float*)alloc((size_t)NNODE * HD * 4);
  // pooling phase (aliases graph scratch; used after graph phase completes)
  off = poolBase;
  f16* final16 = (f16*)alloc((size_t)NNODE * HD * 2);
  f16* g1_16   = (f16*)alloc((size_t)NNODE * GATEH * 2);
  // code phase (aliases pool; used after pooling phase completes)
  off = poolBase;
  f16*   ct16    = (f16*)  alloc((size_t)BTOK * CODED * 2);
  float* tbuf    = (float*)alloc((size_t)BTOK * HD * 4);
  f16*   t16     = (f16*)  alloc((size_t)BTOK * HD * 2);
  f16*   qkv16   = (f16*)  alloc((size_t)BTOK * 3 * HD * 2);
  f16*   q16     = (f16*)  alloc((size_t)NB * NHEAD * LTOK * CHD * 2);
  f16*   k16     = (f16*)  alloc((size_t)NB * NHEAD * LTOK * CHD * 2);
  f16*   vT16    = (f16*)  alloc((size_t)NB * NHEAD * CHD * LTOK * 2);
  float* scores  = (float*)alloc((size_t)NB * NHEAD * LTOK * LTOK * 4);
  f16*   probs16 = (f16*)  alloc((size_t)NB * NHEAD * LTOK * LTOK * 2);
  f16*   o16     = (f16*)  alloc((size_t)BTOK * HD * 2);
  float* tmpadd  = (float*)alloc((size_t)BTOK * HD * 4);
  f16*   ff116   = (f16*)  alloc((size_t)BTOK * 4 * HD * 2);

  auto g1d = [](long long n) { return dim3((unsigned)((n + 255) / 256)); };
  auto cvt = [&](const float* s, f16* d, long long n) {
    k_cvt<<<g1d(n), 256, 0, stream>>>(s, d, n);
  };

  // ---- weight conversion (f32 -> f16) ----
  cvt(Wq,  wq16,  (long long)LCONV * HD * HD);
  cvt(Wk,  wk16,  (long long)LCONV * HD * HD);
  cvt(Wv,  wv16,  (long long)LCONV * HD * HD);
  cvt(We,  we16,  (long long)LCONV * HD * HD);
  cvt(Wsk, wsk16, (long long)LCONV * HD * HD);
  cvt(gW1, gw116, (long long)GATEH * HD);
  cvt(cpW, cpw16, (long long)HD * CODED);
  cvt(aWin, win16, 2LL * 3 * HD * HD);
  cvt(aWout, wout16, 2LL * HD * HD);
  cvt(fW1, fw116, 2LL * 4 * HD * HD);
  cvt(fW2, fw216, 2LL * HD * 4 * HD);
  cvt(hW1, hw116, (long long)NT * HD * 2 * HD);
  cvt(hW2, hw216, (long long)NT * 256 * HD);
  cvt(hW3, hw316, (long long)NT * 128 * 256);

  // ---- embeddings ----
  k_embed_nodes<<<NNODE, 512, 0, stream>>>(x, node_emb, h, h16);
  k_embed_edges<<<NEDGE, 512, 0, stream>>>(edge_attr, edge_emb, ef16);

  // ---- graph TransformerConv stack ----
  const long long NH = (long long)NNODE * HD;
  for (int l = 0; l < LCONV; ++l) {
    size_t wo = (size_t)l * HD * HD;
    gemm(stream, h16, wq16 + wo, bq + l * HD, qb, NNODE, HD, HD, HD, 0,0,0,0,1, 1.f, 0, 0, 1, 1);
    gemm(stream, h16, wk16 + wo, bk + l * HD, kb, NNODE, HD, HD, HD, 0,0,0,0,1, 1.f, 0, 0, 1, 1);
    gemm(stream, h16, wv16 + wo, bv + l * HD, vb, NNODE, HD, HD, HD, 0,0,0,0,1, 1.f, 0, 0, 1, 1);
    gemm(stream, h16, wsk16 + wo, bsk + l * HD, xr, NNODE, HD, HD, HD, 0,0,0,0,1, 1.f, 0, 0, 0, 1);
    gemm(stream, ef16, we16 + wo, (const float*)nullptr, eproj16, NEDGE, HD, HD, HD, 0,0,0,0,1, 1.f, 0, 0, 1, 1);

    k_logits<<<g1d((long long)NEDGE * NHEAD), 256, 0, stream>>>(qb, kb, eproj16, esrc, edst, logits);
    k_init_ms<<<g1d(NNODE * NHEAD), 256, 0, stream>>>(segm, segs, NNODE * NHEAD);
    k_segmax_e<<<g1d((long long)NEDGE * NHEAD), 256, 0, stream>>>(logits, edst, segm);
    k_fixm<<<g1d(NNODE * NHEAD), 256, 0, stream>>>(segm, NNODE * NHEAD);
    k_expsum_e<<<g1d((long long)NEDGE * NHEAD), 256, 0, stream>>>(logits, edst, segm, segs);
    k_zero_f<<<g1d(NH), 256, 0, stream>>>(agg, NH);
    k_scatter_agg<<<g1d((long long)NEDGE * HD), 256, 0, stream>>>(logits, segs, vb, eproj16, esrc, edst, agg);
    k_beta<<<NNODE, 256, 0, stream>>>(agg, xr, Wbeta + (size_t)l * 3 * HD, beta);
    k_zero_d<<<1, 64, 0, stream>>>(red, 2);
    k_hn<<<g1d(NH), 256, 0, stream>>>(agg, xr, beta, hnbuf, red);
    k_update<<<g1d(NH), 256, 0, stream>>>(hnbuf, h, h16, finalb,
                                          lnw + l * HD, lnb + l * HD, red, l == 0 ? 1 : 0);
  }

  // ---- GlobalAttention pooling + scalar context ----
  cvt(finalb, final16, NH);
  gemm(stream, final16, gw116, gb1, g1_16, NNODE, GATEH, HD, GATEH, 0,0,0,0,1, 1.f, 1, 0, 1, 1);
  k_gate2<<<NNODE, 256, 0, stream>>>(g1_16, gW2, gb2, gateb);
  k_init_ms<<<1, 64, 0, stream>>>(pm, ps, NB);
  k_segmax1<<<g1d(NNODE), 256, 0, stream>>>(gateb, batch_vec, pm, NNODE);
  k_fixm<<<1, 64, 0, stream>>>(pm, NB);
  k_expsum1<<<g1d(NNODE), 256, 0, stream>>>(gateb, batch_vec, pm, ps, NNODE);
  k_zero_f<<<g1d(NB * HD), 256, 0, stream>>>(pooled, NB * HD);
  k_pool_scatter<<<g1d(NH), 256, 0, stream>>>(finalb, gateb, ps, batch_vec, pooled);
  k_sctx1<<<g1d(NB * HD), 256, 0, stream>>>(scalars, sW1, sb1, shid);
  k_sctx2<<<g1d(NB * HD), 256, 0, stream>>>(shid, sW2, sb2, pooled, gemb);

  // ---- code branch: projection + 2-layer post-LN transformer ----
  cvt(code_tok, ct16, (long long)BTOK * CODED);
  gemm(stream, ct16, cpw16, cpb, tbuf, BTOK, HD, CODED, HD, 0,0,0,0,1, 1.f, 0, 0, 0, 1);
  cvt(tbuf, t16, (long long)BTOK * HD);
  for (int l = 0; l < 2; ++l) {
    gemm(stream, t16, win16 + (size_t)l * 3 * HD * HD, abin + l * 3 * HD, qkv16,
         BTOK, 3 * HD, HD, 3 * HD, 0,0,0,0,1, 1.f, 0, 0, 1, 1);
    k_qkv_split<<<g1d((long long)BTOK * HD), 256, 0, stream>>>(qkv16, q16, k16, vT16);
    gemm(stream, q16, k16, (const float*)nullptr, scores, LTOK, LTOK, CHD, LTOK,
         (long long)LTOK * CHD, (long long)LTOK * CHD, (long long)LTOK * LTOK, 0, 1,
         0.125f, 0, 0, 0, NB * NHEAD);
    k_softmax<<<NB * NHEAD * LTOK, 256, 0, stream>>>(scores, probs16);
    gemm(stream, probs16, vT16, (const float*)nullptr, o16, LTOK, CHD, LTOK, HD,
         (long long)LTOK * LTOK, (long long)CHD * LTOK,
         (long long)LTOK * HD, (long long)CHD, NHEAD, 1.f, 0, 0, 1, NB * NHEAD);
    gemm(stream, o16, wout16 + (size_t)l * HD * HD, about + l * HD, tmpadd,
         BTOK, HD, HD, HD, 0,0,0,0,1, 1.f, 0, 0, 0, 1);
    k_lnres<<<BTOK, 256, 0, stream>>>(tbuf, tmpadd, t1w + l * HD, t1b + l * HD, t16);
    gemm(stream, t16, fw116 + (size_t)l * 4 * HD * HD, fb1 + l * 4 * HD, ff116,
         BTOK, 4 * HD, HD, 4 * HD, 0,0,0,0,1, 1.f, 2, 0, 1, 1);
    gemm(stream, ff116, fw216 + (size_t)l * HD * 4 * HD, fb2 + l * HD, tmpadd,
         BTOK, HD, 4 * HD, HD, 0,0,0,0,1, 1.f, 0, 0, 0, 1);
    k_lnres<<<BTOK, 256, 0, stream>>>(tbuf, tmpadd, t2w + l * HD, t2b + l * HD, t16);
  }
  k_mean_tokens<<<g1d(NB * HD), 256, 0, stream>>>(tbuf, cemb);

  // ---- fusion + per-target heads ----
  k_fuse<<<g1d(NB * 2 * HD), 256, 0, stream>>>(gemb, cemb, fused16);
  gemm(stream, fused16, hw116, hb1, z1_16, NB, HD, 2 * HD, HD,
       0, (long long)HD * 2 * HD, (long long)NB * HD, 0, 1, 1.f, 1, HD, 1, NT);
  gemm(stream, z1_16, hw216, hb2, z2_16, NB, 256, HD, 256,
       (long long)NB * HD, (long long)256 * HD, (long long)NB * 256, 0, 1, 1.f, 1, 256, 1, NT);
  gemm(stream, z2_16, hw316, hb3, z3_16, NB, 128, 256, 128,
       (long long)NB * 256, (long long)128 * 256, (long long)NB * 128, 0, 1, 1.f, 1, 128, 1, NT);
  k_head4<<<1, 64, 0, stream>>>(z3_16, hW4, hb4, (float*)d_out);
}
